// VAEEncoder_8211977470184
// MI455X (gfx1250) — compile-verified, matrix operand order checked by reading, not run on hardware
//
#include <hip/hip_runtime.h>
#include <hip/hip_bf16.h>
#include <math.h>

// ---------------------------------------------------------------------------
// VAE encoder (2-layer LSTM + mu/logvar projections) for MI455X (gfx1250).
// bf16 WMMA (v_wmma_f32_16x16x32_bf16) with f32 accumulation.
// A-panels staged to LDS via async global->LDS (gfx1250), B streamed with
// prefetch; 4 output tiles (16x64) per wave to amortize fragment loads.
// ---------------------------------------------------------------------------

typedef __attribute__((ext_vector_type(16))) __bf16 v16bf;
typedef __attribute__((ext_vector_type(8)))  __bf16 v8bf;
typedef __attribute__((ext_vector_type(8)))  float  v8f;
typedef __attribute__((ext_vector_type(4)))  int    v4i;

#define SEQL   128
#define BATCH  64
#define EMBD   512
#define HID    512
#define GATES  2048   // 4*HID
#define ZDIM   2048   // NLAYERS*2*HID
#define MROWS  (SEQL*BATCH)  // 8192

#define PANEL_K    512
#define LDS_STRIDE (PANEL_K + 8)   // +16B row pad to spread LDS banks

#if defined(__gfx1250__) && \
    __has_builtin(__builtin_amdgcn_global_load_async_to_lds_b128) && \
    __has_builtin(__builtin_amdgcn_s_wait_asynccnt)
#define USE_ASYNC_LDS 1
#else
#define USE_ASYNC_LDS 0
#endif

// One (A,B,K) accumulation pass: stages A 16xPANEL_K panels into LDS, streams
// B from global, 4 WMMAs per 32-K chunk reusing one A fragment.
__device__ __forceinline__ void gemm_panel_pass(
    const __bf16* __restrict__ A, const __bf16* __restrict__ B,
    __bf16* Als, v8f* acc,
    int m0, int nw0, int K, int nl, int kk)
{
    const __bf16* bp0 = B + (size_t)(nw0 +  0 + nl) * K + kk;
    const __bf16* bp1 = B + (size_t)(nw0 + 16 + nl) * K + kk;
    const __bf16* bp2 = B + (size_t)(nw0 + 32 + nl) * K + kk;
    const __bf16* bp3 = B + (size_t)(nw0 + 48 + nl) * K + kk;

    for (int kp = 0; kp < K; kp += PANEL_K) {
        __syncthreads();                       // previous panel fully consumed
        // ---- cooperatively stage A[m0..m0+16) x [kp..kp+PANEL_K) ----------
        #pragma unroll
        for (int i = 0; i < (16 * PANEL_K) / (256 * 8); ++i) {   // 4 x b128/thread
            const int idx  = (i * 256 + (int)threadIdx.x) * 8;
            const int row  = idx / PANEL_K;
            const int colp = idx % PANEL_K;
            const __bf16* g = A + (size_t)(m0 + row) * K + kp + colp;
            __bf16*       l = Als + row * LDS_STRIDE + colp;
#if USE_ASYNC_LDS
            __builtin_amdgcn_global_load_async_to_lds_b128(
                (__attribute__((address_space(1))) v4i*)g,
                (__attribute__((address_space(3))) v4i*)l, 0, 0);
#else
            *(v8bf*)l = *(const v8bf*)g;
#endif
        }
#if USE_ASYNC_LDS
        __builtin_amdgcn_s_wait_asynccnt(0);
#endif
        __syncthreads();                       // panel visible to all waves

        // ---- 16 chunks of K=32: 2 ds_load + 8 global_load + 4 WMMA --------
        #pragma unroll 2
        for (int k = 0; k < PANEL_K; k += 32) {
            const int kg = kp + k;
            const __bf16* lrow = Als + nl * LDS_STRIDE + k + kk;
            v8bf alo = *(const v8bf*)(lrow);
            v8bf ahi = *(const v8bf*)(lrow + 16);
            v16bf af;
            #pragma unroll
            for (int i = 0; i < 8; ++i) { af[i] = alo[i]; af[i + 8] = ahi[i]; }

            __builtin_prefetch(bp0 + kg + PANEL_K, 0, 1);   // stream-ahead on B

            #pragma unroll
            for (int t = 0; t < 4; ++t) {
                const __bf16* bp = (t == 0 ? bp0 : t == 1 ? bp1 : t == 2 ? bp2 : bp3) + kg;
                v8bf blo = *(const v8bf*)(bp);
                v8bf bhi = *(const v8bf*)(bp + 16);
                v16bf bfr;
                #pragma unroll
                for (int i = 0; i < 8; ++i) { bfr[i] = blo[i]; bfr[i + 8] = bhi[i]; }
                acc[t] = __builtin_amdgcn_wmma_f32_16x16x32_bf16(
                             false, af, false, bfr, (short)0, acc[t], false, false);
            }
        }
    }
}

// ---------------------------------------------------------------------------
// D[M,N] = A[M,K]*B[N,K]^T (+ A2[M,K2]*B2[N,K2]^T) (+bias[N]) (+Cinit[M,N])
// Block: 256 threads = 8 waves; block tile = 16(M) x 512(N); wave = 16 x 64.
// grid = (N/512, M/16). N,K,K2 multiples of 512; M multiple of 16 (no guards:
// every thread must reach the barriers, and EXEC is all-1s at each WMMA).
// ---------------------------------------------------------------------------
__global__ __launch_bounds__(256)
void k_gemm_wmma(const __bf16* __restrict__ A,  const __bf16* __restrict__ B,
                 const __bf16* __restrict__ A2, const __bf16* __restrict__ B2,
                 const float*  __restrict__ bias, const float* __restrict__ Cinit,
                 float* __restrict__ D, __bf16* __restrict__ Dbf,
                 int M, int N, int K, int K2)
{
    __shared__ __align__(16) __bf16 Als[16 * LDS_STRIDE];

    const int lane = threadIdx.x & 31;
    const int wave = threadIdx.x >> 5;
    const int nl   = lane & 15;
    const int kk   = (lane < 16) ? 0 : 8;            // A/B K-half per lane group
    const int m0   = blockIdx.y * 16;
    const int nw0  = blockIdx.x * 512 + wave * 64;   // wave's 4 N-tiles
    const int rowbase = m0 + ((lane >> 4) << 3);     // C/D: M = vgpr (+8 hi lanes)

    v8f acc[4] = {{}, {}, {}, {}};
    if (Cinit) {
        #pragma unroll
        for (int t = 0; t < 4; ++t)
            #pragma unroll
            for (int r = 0; r < 8; ++r)
                acc[t][r] = Cinit[(size_t)(rowbase + r) * N + nw0 + t * 16 + nl];
    }
    if (bias) {
        #pragma unroll
        for (int t = 0; t < 4; ++t) {
            const float bv = bias[nw0 + t * 16 + nl];
            #pragma unroll
            for (int r = 0; r < 8; ++r) acc[t][r] += bv;
        }
    }

    gemm_panel_pass(A, B, Als, acc, m0, nw0, K, nl, kk);
    if (A2)
        gemm_panel_pass(A2, B2, Als, acc, m0, nw0, K2, nl, kk);

    #pragma unroll
    for (int t = 0; t < 4; ++t) {
        const int col = nw0 + t * 16 + nl;
        #pragma unroll
        for (int r = 0; r < 8; ++r) {
            const size_t idx = (size_t)(rowbase + r) * N + col;
            const float v = acc[t][r];
            if (D)   D[idx]   = v;
            if (Dbf) Dbf[idx] = (__bf16)v;
        }
    }
}

// ---------------------------------------------------------------------------
// Elementwise helpers
// ---------------------------------------------------------------------------
__device__ __forceinline__ float sigf(float x) { return 1.f / (1.f + __expf(-x)); }

// LSTM cell: gates[64,2048] (i,f,g,o) -> c (f32 state, in-place), h (bf16 state),
// and writes (h,c) as bf16 into the zipped buffer row for this step/layer.
__global__ void k_lstm_cell(const float* __restrict__ gates, float* __restrict__ c,
                            __bf16* __restrict__ h_bf, __bf16* __restrict__ zip_row)
{
    const int tid = blockIdx.x * blockDim.x + threadIdx.x;
    if (tid >= BATCH * HID) return;
    const int b = tid >> 9;          // /512
    const int j = tid & 511;
    const float* g = gates + (size_t)b * GATES;
    const float gi = g[j], gf = g[512 + j], gg = g[1024 + j], go = g[1536 + j];
    const float cn = sigf(gf) * c[tid] + sigf(gi) * tanhf(gg);
    const float hn = sigf(go) * tanhf(cn);
    c[tid]    = cn;
    h_bf[tid] = (__bf16)hn;
    zip_row[(size_t)b * ZDIM + j]       = (__bf16)hn;
    zip_row[(size_t)b * ZDIM + 512 + j] = (__bf16)cn;
}

__global__ void k_f32_to_bf16(const float* __restrict__ in, __bf16* __restrict__ out, int n)
{
    const int i = blockIdx.x * blockDim.x + threadIdx.x;
    if (i < n) out[i] = (__bf16)in[i];
}

__global__ void k_add2(const float* __restrict__ a, const float* __restrict__ b,
                       float* __restrict__ o, int n)
{
    const int i = blockIdx.x * blockDim.x + threadIdx.x;
    if (i < n) o[i] = a[i] + b[i];
}

__global__ void k_zero_u32(unsigned* __restrict__ p, int nwords)
{
    const int i = blockIdx.x * blockDim.x + threadIdx.x;
    if (i < nwords) p[i] = 0u;
}

// Embedding gather -> bf16 activation rows [MROWS, EMBD]; row = s*BATCH + b.
__global__ void k_embed(const int* __restrict__ src, const float* __restrict__ emb,
                        __bf16* __restrict__ A)
{
    const int row = blockIdx.x;
    const int tok = src[row];
    const float* e = emb + (size_t)tok * EMBD;
    for (int j = threadIdx.x; j < EMBD; j += blockDim.x)
        A[(size_t)row * EMBD + j] = (__bf16)e[j];
}

__global__ void k_senlen(const int* __restrict__ src, float* __restrict__ out)
{
    const int b = threadIdx.x;
    if (b < BATCH) {
        int cnt = 0;
        for (int s = 0; s < SEQL; ++s) cnt += (src[s * BATCH + b] > 0) ? 1 : 0;
        out[b] = (float)cnt;
    }
}

__global__ void k_copy_f32(const float* __restrict__ in, float* __restrict__ out, int n)
{
    const int i = blockIdx.x * blockDim.x + threadIdx.x;
    if (i < n) out[i] = in[i];
}

// ---------------------------------------------------------------------------
// Launch
// ---------------------------------------------------------------------------
extern "C" void kernel_launch(void* const* d_in, const int* in_sizes, int n_in,
                              void* d_out, int out_size, void* d_ws, size_t ws_size,
                              hipStream_t stream)
{
    (void)in_sizes; (void)n_in; (void)out_size; (void)ws_size;

    const int*   src  = (const int*)  d_in[0];
    const float* emb  = (const float*)d_in[1];
    const float* Wih0 = (const float*)d_in[2];
    const float* Whh0 = (const float*)d_in[3];
    const float* bih0 = (const float*)d_in[4];
    const float* bhh0 = (const float*)d_in[5];
    const float* Wih1 = (const float*)d_in[6];
    const float* Whh1 = (const float*)d_in[7];
    const float* bih1 = (const float*)d_in[8];
    const float* bhh1 = (const float*)d_in[9];
    const float* Wmu  = (const float*)d_in[10];
    const float* bmu  = (const float*)d_in[11];
    const float* Wvar = (const float*)d_in[12];
    const float* bvar = (const float*)d_in[13];

    float* out_h   = (float*)d_out;                          // [64, 2048]
    float* out_mu  = out_h  + (size_t)BATCH * ZDIM;          // [8192, 2048]
    float* out_lv  = out_mu + (size_t)MROWS * ZDIM;          // [8192, 2048]
    float* out_len = out_lv + (size_t)MROWS * ZDIM;          // [64]

    // ---- workspace carve-up -------------------------------------------------
    char* wptr = (char*)d_ws;
    auto carve = [&](size_t bytes) -> char* {
        char* p = wptr;
        wptr += (bytes + 255) & ~(size_t)255;
        return p;
    };
    __bf16* Wih0b = (__bf16*)carve((size_t)GATES * EMBD * 2);
    __bf16* Whh0b = (__bf16*)carve((size_t)GATES * HID  * 2);
    __bf16* Wih1b = (__bf16*)carve((size_t)GATES * HID  * 2);
    __bf16* Whh1b = (__bf16*)carve((size_t)GATES * HID  * 2);
    __bf16* Wmub  = (__bf16*)carve((size_t)ZDIM  * ZDIM * 2);
    __bf16* Wvarb = (__bf16*)carve((size_t)ZDIM  * ZDIM * 2);
    __bf16* Aemb  = (__bf16*)carve((size_t)MROWS * EMBD * 2);
    float*  X0    = (float*) carve((size_t)MROWS * GATES * 4);   // x@Wih0^T + biases
    __bf16* Zip   = (__bf16*)carve((size_t)MROWS * ZDIM * 2);    // [S*B, h0|c0|h1|c1]
    float*  g0    = (float*) carve((size_t)BATCH * GATES * 4);
    float*  g1    = (float*) carve((size_t)BATCH * GATES * 4);
    __bf16* h0b   = (__bf16*)carve((size_t)BATCH * HID * 2);
    __bf16* h1b   = (__bf16*)carve((size_t)BATCH * HID * 2);
    float*  c0    = (float*) carve((size_t)BATCH * HID * 4);
    float*  c1    = (float*) carve((size_t)BATCH * HID * 4);
    float*  bsum0 = (float*) carve((size_t)GATES * 4);
    float*  bsum1 = (float*) carve((size_t)GATES * 4);

    auto cdiv = [](int a, int b) { return (a + b - 1) / b; };

    // ---- one-time (per launch) prep ----------------------------------------
    k_f32_to_bf16<<<cdiv(GATES*EMBD,256),256,0,stream>>>(Wih0, Wih0b, GATES*EMBD);
    k_f32_to_bf16<<<cdiv(GATES*HID ,256),256,0,stream>>>(Whh0, Whh0b, GATES*HID);
    k_f32_to_bf16<<<cdiv(GATES*HID ,256),256,0,stream>>>(Wih1, Wih1b, GATES*HID);
    k_f32_to_bf16<<<cdiv(GATES*HID ,256),256,0,stream>>>(Whh1, Whh1b, GATES*HID);
    k_f32_to_bf16<<<cdiv(ZDIM*ZDIM ,256),256,0,stream>>>(Wmu,  Wmub,  ZDIM*ZDIM);
    k_f32_to_bf16<<<cdiv(ZDIM*ZDIM ,256),256,0,stream>>>(Wvar, Wvarb, ZDIM*ZDIM);
    k_add2<<<cdiv(GATES,256),256,0,stream>>>(bih0, bhh0, bsum0, GATES);
    k_add2<<<cdiv(GATES,256),256,0,stream>>>(bih1, bhh1, bsum1, GATES);

    k_zero_u32<<<cdiv(BATCH*HID,256),256,0,stream>>>((unsigned*)c0,  BATCH*HID);
    k_zero_u32<<<cdiv(BATCH*HID,256),256,0,stream>>>((unsigned*)c1,  BATCH*HID);
    k_zero_u32<<<cdiv(BATCH*HID/2,256),256,0,stream>>>((unsigned*)h0b, BATCH*HID/2);
    k_zero_u32<<<cdiv(BATCH*HID/2,256),256,0,stream>>>((unsigned*)h1b, BATCH*HID/2);

    k_embed<<<MROWS,128,0,stream>>>(src, emb, Aemb);
    k_senlen<<<1,64,0,stream>>>(src, out_len);

    // ---- hoisted layer-0 input projection: X0 = Aemb @ Wih0^T + (bih0+bhh0) --
    {
        dim3 g(GATES/512, MROWS/16);
        k_gemm_wmma<<<g,256,0,stream>>>(Aemb, Wih0b, nullptr, nullptr,
                                        bsum0, nullptr, X0, nullptr,
                                        MROWS, GATES, EMBD, 0);
    }

    // ---- sequential LSTM scan ----------------------------------------------
    dim3 gstep(GATES/512, BATCH/16);
    const int cellBlocks = cdiv(BATCH*HID, 256);
    for (int s = 0; s < SEQL; ++s) {
        __bf16* zrow = Zip + (size_t)s * BATCH * ZDIM;
        // layer 0: gates = X0[s] + h0 @ Whh0^T
        k_gemm_wmma<<<gstep,256,0,stream>>>(h0b, Whh0b, nullptr, nullptr,
                                            nullptr, X0 + (size_t)s*BATCH*GATES,
                                            g0, nullptr, BATCH, GATES, HID, 0);
        k_lstm_cell<<<cellBlocks,256,0,stream>>>(g0, c0, h0b, zrow);           // h0|c0
        // layer 1: gates = h0n @ Wih1^T + h1 @ Whh1^T + (bih1+bhh1)  (fused)
        k_gemm_wmma<<<gstep,256,0,stream>>>(h0b, Wih1b, h1b, Whh1b,
                                            bsum1, nullptr,
                                            g1, nullptr, BATCH, GATES, HID, HID);
        k_lstm_cell<<<cellBlocks,256,0,stream>>>(g1, c1, h1b, zrow + 1024);    // h1|c1
    }

    // ---- mu / logvar projections -------------------------------------------
    {
        dim3 g(ZDIM/512, MROWS/16);
        k_gemm_wmma<<<g,256,0,stream>>>(Zip, Wmub, nullptr, nullptr,
                                        bmu, nullptr, out_mu, nullptr,
                                        MROWS, ZDIM, ZDIM, 0);
        k_gemm_wmma<<<g,256,0,stream>>>(Zip, Wvarb, nullptr, nullptr,
                                        bvar, nullptr, out_lv, nullptr,
                                        MROWS, ZDIM, ZDIM, 0);
    }

    // ---- h_out = mu[last step] ---------------------------------------------
    k_copy_f32<<<cdiv(BATCH*ZDIM,256),256,0,stream>>>(
        out_mu + (size_t)(SEQL-1)*BATCH*ZDIM, out_h, BATCH*ZDIM);
}
